// MaskAdaptiveLogSoftmax_56908316672215
// MI455X (gfx1250) — matrix operand order repaired and checked by reading, not compiled, not run on hardware
//
#include <hip/hip_runtime.h>
#include <math.h>

typedef __attribute__((ext_vector_type(16))) __bf16 v16bf;
typedef __attribute__((ext_vector_type(8)))  float  v8f;

#define D_DIM   256
#define VOCAB   50000
#define NTOK    2048
#define BM      64      // workgroup rows (tokens)
#define BN      128     // workgroup cols (vocab)
#define KC      64      // K chunk staged in LDS
#define NCT     391     // ceil(50000 / 128) column tiles

// ---- f32 -> bf16 (RNE) and back, by bit manipulation ----
__device__ __forceinline__ unsigned short f32_to_bf16(float f) {
    union { float f; unsigned int u; } v; v.f = f;
    unsigned int u = v.u;
    unsigned int r = (u + 0x7FFFu + ((u >> 16) & 1u)) >> 16;
    return (unsigned short)r;
}
__device__ __forceinline__ float bf16_to_f32(unsigned short h) {
    union { unsigned int u; float f; } v; v.u = ((unsigned int)h) << 16;
    return v.f;
}
// split one float4 into packed hi (uint2) and lo (uint2) bf16 pairs
__device__ __forceinline__ void split4(float4 v, uint2* hi, uint2* lo) {
    unsigned short h0 = f32_to_bf16(v.x), h1 = f32_to_bf16(v.y),
                   h2 = f32_to_bf16(v.z), h3 = f32_to_bf16(v.w);
    unsigned short l0 = f32_to_bf16(v.x - bf16_to_f32(h0));
    unsigned short l1 = f32_to_bf16(v.y - bf16_to_f32(h1));
    unsigned short l2 = f32_to_bf16(v.z - bf16_to_f32(h2));
    unsigned short l3 = f32_to_bf16(v.w - bf16_to_f32(h3));
    hi->x = (unsigned)h0 | ((unsigned)h1 << 16);
    hi->y = (unsigned)h2 | ((unsigned)h3 << 16);
    lo->x = (unsigned)l0 | ((unsigned)l1 << 16);
    lo->y = (unsigned)l2 | ((unsigned)l3 << 16);
}

union FragBF16 { v16bf v; int4 q[2]; };

// ------------------------------------------------------------------
// Kernel 1: tiled GEMM (bf16-split, fp32-accurate) + bias + cluster
// mask + per-(token, col-tile) partial sum(exp(z)).  z -> d_out.
// Each wave computes a 32x32 block (2x2 WMMA tiles, 12 WMMA / K-step).
// ------------------------------------------------------------------
__global__ __launch_bounds__(256)
void gemm_logits_kernel(const float* __restrict__ x,
                        const int*   __restrict__ targets,
                        const float* __restrict__ w0,
                        const float* __restrict__ w1,
                        const float* __restrict__ w2,
                        const float* __restrict__ b0,
                        const float* __restrict__ b1,
                        const float* __restrict__ b2,
                        float* __restrict__ out,
                        float* __restrict__ partials)
{
    const int tid  = threadIdx.x;
    const int lane = tid & 31;
    const int wave = tid >> 5;
    const int half = lane >> 4;
    const int l16  = lane & 15;
    const int rb   = wave >> 2;      // 0..1 : 32-row block
    const int cb   = wave & 3;       // 0..3 : 32-col block
    const int M0   = blockIdx.x * BM;
    const int N0   = blockIdx.y * BN;

    __shared__ unsigned short sXhi[BM][KC];   // 8 KB
    __shared__ unsigned short sXlo[BM][KC];   // 8 KB
    __shared__ unsigned short sWhi[BN][KC];   // 16 KB
    __shared__ unsigned short sWlo[BN][KC];   // 16 KB
    __shared__ float sSums[4][BM];
    __shared__ int   sTgt[BM];

    if (tid < BM) sTgt[tid] = targets[M0 + tid];

    v8f acc[2][2];
    #pragma unroll
    for (int i = 0; i < 2; ++i)
        #pragma unroll
        for (int j = 0; j < 2; ++j)
            acc[i][j] = (v8f){0.f,0.f,0.f,0.f,0.f,0.f,0.f,0.f};

    for (int kc = 0; kc < D_DIM; kc += KC) {
        __syncthreads();   // protect LDS reuse across chunks (covers sTgt on first pass)

        // ---- batched, branchless loads: X chunk (64x64 = 1024 float4) ----
        float4 xv[4];
        #pragma unroll
        for (int it = 0; it < 4; ++it) {
            int i = tid + 256 * it;              // < 1024
            int row = i >> 4, c4 = i & 15;
            xv[it] = *(const float4*)(x + (size_t)(M0 + row) * D_DIM + kc + c4 * 4);
        }
        // ---- batched, branchless loads: W chunk (128x64 = 2048 float4) ----
        float4 wv[8];
        #pragma unroll
        for (int it = 0; it < 8; ++it) {
            int i = tid + 256 * it;              // < 2048
            int row = i >> 4, c4 = i & 15;
            int n = N0 + row;
            int nc = (n < VOCAB) ? n : (VOCAB - 1);          // clamp (branchless)
            const float* base = (nc < 20000) ? w0 : ((nc < 40000) ? w1 : w2);
            int rel = (nc < 20000) ? nc : ((nc < 40000) ? nc - 20000 : nc - 40000);
            float4 v = *(const float4*)(base + (size_t)rel * D_DIM + kc + c4 * 4);
            float msk = (n < VOCAB) ? 1.f : 0.f;
            v.x *= msk; v.y *= msk; v.z *= msk; v.w *= msk;
            wv[it] = v;
        }
        // prefetch next K chunk (global_prefetch_b8)
        if (kc + KC < D_DIM) {
            int row = tid >> 4, c4 = tid & 15;
            __builtin_prefetch(x + (size_t)(M0 + row) * D_DIM + kc + KC + c4 * 4, 0, 0);
        }

        // ---- convert + pack + LDS store (b64) ----
        #pragma unroll
        for (int it = 0; it < 4; ++it) {
            int i = tid + 256 * it;
            int row = i >> 4, c4 = i & 15;
            uint2 hi, lo;
            split4(xv[it], &hi, &lo);
            *(uint2*)&sXhi[row][c4 * 4] = hi;
            *(uint2*)&sXlo[row][c4 * 4] = lo;
        }
        #pragma unroll
        for (int it = 0; it < 8; ++it) {
            int i = tid + 256 * it;
            int row = i >> 4, c4 = i & 15;
            uint2 hi, lo;
            split4(wv[it], &hi, &lo);
            *(uint2*)&sWhi[row][c4 * 4] = hi;
            *(uint2*)&sWlo[row][c4 * 4] = lo;
        }
        __syncthreads();

        #pragma unroll
        for (int k0 = 0; k0 < KC; k0 += 32) {
            FragBF16 ahi[2], alo[2], bhi[2], blo[2];
            #pragma unroll
            for (int rt = 0; rt < 2; ++rt) {
                const unsigned short* ph = &sXhi[rb * 32 + rt * 16 + l16][0];
                const unsigned short* pl = &sXlo[rb * 32 + rt * 16 + l16][0];
                // A frag (16x32, M=l16): K = k0 + 8*half + {0..7}, then +16
                ahi[rt].q[0] = *(const int4*)(ph + k0 + 8 * half);
                ahi[rt].q[1] = *(const int4*)(ph + k0 + 8 * half + 16);
                alo[rt].q[0] = *(const int4*)(pl + k0 + 8 * half);
                alo[rt].q[1] = *(const int4*)(pl + k0 + 8 * half + 16);
            }
            #pragma unroll
            for (int ct = 0; ct < 2; ++ct) {
                const unsigned short* ph = &sWhi[cb * 32 + ct * 16 + l16][0];
                const unsigned short* pl = &sWlo[cb * 32 + ct * 16 + l16][0];
                // B frag (32x16, N=l16): K = k0 + 8*half + {0..7}, then +16
                bhi[ct].q[0] = *(const int4*)(ph + k0 + 8 * half);
                bhi[ct].q[1] = *(const int4*)(ph + k0 + 8 * half + 16);
                blo[ct].q[0] = *(const int4*)(pl + k0 + 8 * half);
                blo[ct].q[1] = *(const int4*)(pl + k0 + 8 * half + 16);
            }
            #pragma unroll
            for (int rt = 0; rt < 2; ++rt)
                #pragma unroll
                for (int ct = 0; ct < 2; ++ct) {
                    // fp32-accurate product: hi*hi + hi*lo + lo*hi
                    acc[rt][ct] = __builtin_amdgcn_wmma_f32_16x16x32_bf16(
                        false, ahi[rt].v, false, bhi[ct].v, (short)0, acc[rt][ct], false, false);
                    acc[rt][ct] = __builtin_amdgcn_wmma_f32_16x16x32_bf16(
                        false, ahi[rt].v, false, blo[ct].v, (short)0, acc[rt][ct], false, false);
                    acc[rt][ct] = __builtin_amdgcn_wmma_f32_16x16x32_bf16(
                        false, alo[rt].v, false, bhi[ct].v, (short)0, acc[rt][ct], false, false);
                }
        }
    }

    // ---- epilogue: bias + mask, store z, per-row sum(exp(z)) ----
    int  ncol[2];  bool nval[2];  int cl_l[2], cl_r[2];  float bias[2];
    #pragma unroll
    for (int ct = 0; ct < 2; ++ct) {
        int n = N0 + cb * 32 + ct * 16 + l16;
        ncol[ct] = n;
        nval[ct] = (n < VOCAB);
        int nc = nval[ct] ? n : (VOCAB - 1);                 // clamp (branchless loads)
        const float* bb = (nc < 20000) ? b0 : ((nc < 40000) ? b1 : b2);
        int rel = (nc < 20000) ? nc : ((nc < 40000) ? nc - 20000 : nc - 40000);
        cl_l[ct] = (nc < 20000) ? 0     : ((nc < 40000) ? 20000 : 40000);
        cl_r[ct] = (nc < 20000) ? 20000 : ((nc < 40000) ? 40000 : VOCAB);
        bias[ct] = bb[rel];
    }

    #pragma unroll
    for (int rt = 0; rt < 2; ++rt) {
        #pragma unroll
        for (int r = 0; r < 8; ++r) {
            const int mloc = rb * 32 + rt * 16 + 8 * half + r;
            const int t = sTgt[mloc];
            float esum = 0.f;
            #pragma unroll
            for (int ct = 0; ct < 2; ++ct) {
                float z = 0.f;
                if (nval[ct] && t >= cl_l[ct] && t < cl_r[ct])
                    z = acc[rt][ct][r] + bias[ct];
                if (nval[ct])
                    out[(size_t)(M0 + mloc) * VOCAB + ncol[ct]] = z;
                esum += nval[ct] ? expf(z) : 0.f;   // out-of-cluster z=0 -> exp=1
            }
            // reduce over the 16 lanes of this half (32 columns of the wave)
            for (int off = 1; off < 16; off <<= 1)
                esum += __shfl_xor(esum, off, 32);
            if (l16 == 0) sSums[cb][mloc] = esum;
        }
    }
    __syncthreads();
    if (tid < BM) {
        // fixed summation order over the 4 column blocks -> deterministic
        float s = sSums[0][tid] + sSums[1][tid] + sSums[2][tid] + sSums[3][tid];
        partials[(size_t)(M0 + tid) * NCT + blockIdx.y] = s;
    }
}

// ------------------------------------------------------------------
// Kernel 2: per-token deterministic reduction -> lse = log(sum(exp))
// ------------------------------------------------------------------
__global__ __launch_bounds__(256)
void reduce_lse_kernel(const float* __restrict__ partials,
                       float* __restrict__ lse)
{
    const int m = blockIdx.x;
    __shared__ float red[256];
    float s = 0.f;
    for (int i = threadIdx.x; i < NCT; i += 256)
        s += partials[(size_t)m * NCT + i];
    red[threadIdx.x] = s;
    __syncthreads();
    for (int st = 128; st > 0; st >>= 1) {
        if (threadIdx.x < st) red[threadIdx.x] += red[threadIdx.x + st];
        __syncthreads();
    }
    if (threadIdx.x == 0) lse[m] = logf(red[0]);
}

// ------------------------------------------------------------------
// Kernel 3: streaming float4 pass, out = z - lse  (in place)
// ------------------------------------------------------------------
__global__ __launch_bounds__(256)
void sub_lse_kernel(float* __restrict__ out,
                    const float* __restrict__ lse)
{
    const int m  = blockIdx.y;
    const int i4 = blockIdx.x * 256 + threadIdx.x;       // float4 index in row
    if (i4 * 4 < VOCAB) {
        const float L = lse[m];
        float4* p = (float4*)(out + (size_t)m * VOCAB) + i4;
        float4 v = *p;
        v.x -= L; v.y -= L; v.z -= L; v.w -= L;
        *p = v;
    }
}

extern "C" void kernel_launch(void* const* d_in, const int* in_sizes, int n_in,
                              void* d_out, int out_size, void* d_ws, size_t ws_size,
                              hipStream_t stream) {
    const float* x       = (const float*)d_in[0];
    const int*   targets = (const int*)  d_in[1];
    const float* w0      = (const float*)d_in[2];
    const float* w1      = (const float*)d_in[3];
    const float* w2      = (const float*)d_in[4];
    const float* b0      = (const float*)d_in[5];
    const float* b1      = (const float*)d_in[6];
    const float* b2      = (const float*)d_in[7];
    float* out = (float*)d_out;

    float* partials = (float*)d_ws;                       // [NTOK][NCT] ~ 3.2 MB
    float* lse      = partials + (size_t)NTOK * NCT;      // [NTOK]

    dim3 g1(NTOK / BM, NCT);                              // 32 x 391
    gemm_logits_kernel<<<g1, 256, 0, stream>>>(x, targets, w0, w1, w2,
                                               b0, b1, b2, out, partials);

    reduce_lse_kernel<<<NTOK, 256, 0, stream>>>(partials, lse);

    dim3 g3((VOCAB / 4 + 255) / 256, NTOK);               // 49 x 2048
    sub_lse_kernel<<<g3, 256, 0, stream>>>(out, lse);
}
